// SemiSupervisedNetwork_867583393952
// MI455X (gfx1250) — matrix operand order repaired and checked by reading, not compiled, not run on hardware
//
#include <hip/hip_runtime.h>

// ---------------------------------------------------------------------------
// LIF spiking network on MI455X (gfx1250, wave32, WMMA, TDM + async-to-LDS).
//
//   K1 transpose_input : (256,784,250) f32 -> xT (250,256,832) f16 (K padded)
//   K2 transpose_w1    : relu(W1) (784,1024) f32 -> W1T (1024,832) f16
//   K3 prep_w2         : relu(W2) (1024,10) f32 -> W2T (16,1024) f16
//   KA lif_hidden      : per (b,h) tile, loop t=0..249:
//                          I_h = x_t @ W1 via v_wmma_f32_16x16x32_f16;
//                          x K-chunks staged by TENSOR_LOAD_TO_LDS (wave 0,
//                          TENSORcnt) with GLOBAL_LOAD_ASYNC_TO_LDS fallback;
//                          v_h LIF update carried in registers;
//                          spike -> sH (250,256,1024) f16
//   KB lif_output      : per 16-batch tile, loop t: I_o = sH[t-1] @ W2 (WMMA),
//                          LIF update, spikes (256,10,250) + rates (256,10)
//
// Workspace: xT 106.5MB | W1T 1.7MB | W2T 32KB | sH 131MB  => ~240MB.
// ---------------------------------------------------------------------------

typedef __attribute__((ext_vector_type(16))) _Float16     v16h;
typedef __attribute__((ext_vector_type(8)))  _Float16     v8h;
typedef __attribute__((ext_vector_type(8)))  float        v8f;
typedef __attribute__((ext_vector_type(4)))  int          v4i;
typedef __attribute__((ext_vector_type(4)))  unsigned int u32x4;
typedef __attribute__((ext_vector_type(8)))  int          i32x8;
typedef __attribute__((ext_vector_type(4)))  int          i32x4;

#define AS1 __attribute__((address_space(1)))
#define AS3 __attribute__((address_space(3)))

#define B_    256
#define NIN   784
#define K_    832      // NIN padded to 13*64
#define H_    1024
#define NOUT  10
#define T_    250

// hidden kernel tiling
#define BT    64       // batch rows per workgroup
#define HT    64       // hidden cols per workgroup
#define KC    64       // K-chunk per pipeline stage (2 WMMA K-slices)
#define SA    72       // LDS A row stride in halves (144B -> conflict-free)
#define SB    840      // LDS W row stride in halves (1680B -> conflict-free)
#define NKB   (K_ / KC)   // 13

#if defined(__gfx1250__) &&                                                    \
    __has_builtin(__builtin_amdgcn_global_load_async_to_lds_b128) &&           \
    __has_builtin(__builtin_amdgcn_s_wait_asynccnt)
#define HAVE_ASYNC_LDS 1
#else
#define HAVE_ASYNC_LDS 0
#endif

#if defined(__gfx1250__) &&                                                    \
    __has_builtin(__builtin_amdgcn_tensor_load_to_lds) &&                      \
    __has_builtin(__builtin_amdgcn_s_wait_tensorcnt)
#define HAVE_TDM 1
#if __has_include(<hip/amd_detail/amd_gfx1250_TDM.h>)
#define TDM_ARITY 6    // therock / clang-23 header lane
#else
#define TDM_ARITY 5    // ROCm 7.2 / clang-22 lane
#endif
#else
#define HAVE_TDM 0
#endif

union Frag { v16h v; v8h h[2]; };

// ------------------------- pre-pass kernels --------------------------------

__global__ __launch_bounds__(256) void transpose_input(
    const float* __restrict__ x,    // (256, 784, 250)
    _Float16* __restrict__ xT)      // (250, 256, 832)
{
  __shared__ float tile[32][33];
  const int b  = blockIdx.x;
  const int i0 = blockIdx.y * 32;
  const int t0 = blockIdx.z * 32;
  const int tx = threadIdx.x & 31;
  const int ty = threadIdx.x >> 5;   // 0..7
#pragma unroll
  for (int r = 0; r < 4; ++r) {
    int i = i0 + ty + r * 8;
    int t = t0 + tx;
    float v = 0.0f;
    if (i < NIN && t < T_) v = x[((size_t)b * NIN + i) * T_ + t];
    tile[ty + r * 8][tx] = v;        // rows >= 784 become zero padding
  }
  __syncthreads();
#pragma unroll
  for (int r = 0; r < 4; ++r) {
    int t = t0 + ty + r * 8;
    int i = i0 + tx;                 // i0 max 800 -> i < 832 always
    if (t < T_)
      xT[((size_t)t * B_ + b) * K_ + i] = (_Float16)tile[tx][ty + r * 8];
  }
}

__global__ __launch_bounds__(256) void transpose_w1(
    const float* __restrict__ w1,   // (784, 1024)
    _Float16* __restrict__ W1T)     // (1024, 832), relu applied
{
  __shared__ float tile[32][33];
  const int k0 = blockIdx.x * 32;   // 26 blocks
  const int h0 = blockIdx.y * 32;   // 32 blocks
  const int tx = threadIdx.x & 31;
  const int ty = threadIdx.x >> 5;
#pragma unroll
  for (int r = 0; r < 4; ++r) {
    int k = k0 + ty + r * 8;
    float v = (k < NIN) ? w1[(size_t)k * H_ + (h0 + tx)] : 0.0f;
    tile[ty + r * 8][tx] = fmaxf(v, 0.0f);
  }
  __syncthreads();
#pragma unroll
  for (int r = 0; r < 4; ++r) {
    int h = h0 + ty + r * 8;
    int k = k0 + tx;
    W1T[(size_t)h * K_ + k] = (_Float16)tile[tx][ty + r * 8];
  }
}

__global__ __launch_bounds__(256) void prep_w2(
    const float* __restrict__ w2,   // (1024, 10)
    _Float16* __restrict__ W2T)     // (16, 1024), relu, cols 10..15 zero
{
  int idx = blockIdx.x * 256 + threadIdx.x;   // 16*1024 elements
  if (idx >= 16 * H_) return;
  int o = idx >> 10;
  int h = idx & (H_ - 1);
  float v = (o < NOUT) ? fmaxf(w2[(size_t)h * NOUT + o], 0.0f) : 0.0f;
  W2T[idx] = (_Float16)v;
}

// ------------------------- hidden layer ------------------------------------

__device__ __forceinline__ unsigned lds_addr_of(const void* p) {
  return (unsigned)(uintptr_t)(AS3 const void*)p;
}

#if HAVE_TDM
// One TDM descriptor moves a BTxKC f16 tile (global row stride K_ elements)
// into LDS with +4 DWORD padding after each 32-DWORD row => LDS stride SA.
__device__ __forceinline__ void tdm_stage_chunk(const _Float16* gsrc,
                                                unsigned lds_byte_addr) {
  unsigned long long ga = (unsigned long long)(uintptr_t)gsrc;
  u32x4 g0;
  g0[0] = 1u;                                   // count=1 user descriptor
  g0[1] = lds_byte_addr;                        // lds_addr (bytes)
  g0[2] = (unsigned)(ga & 0xffffffffu);         // global_addr[31:0]
  g0[3] = (unsigned)((ga >> 32) & 0x01ffffffu)  // global_addr[56:32]
          | (2u << 30);                         // type = 2 ("image")
  i32x8 g1;
  g1[0] = (1 << 16)      // data_size = 2 bytes
        | (1 << 20)      // pad_enable
        | (4 << 22)      // pad_interval: 2^(4+1) = 32 DWORDs (= one 64-half row)
        | (3 << 25);     // pad_amount: 4 DWORDs (= 8 halves -> stride SA)
  g1[1] = (int)((unsigned)K_ << 16);   // tensor_dim0 = 832 (bits 79:48 low part)
  g1[2] = (int)(64u << 16);            // tensor_dim0 hi=0 | tensor_dim1 = 64
  g1[3] = (int)((unsigned)KC << 16);   // tensor_dim1 hi=0 | tile_dim0 = 64
  g1[4] = BT;                          // tile_dim1 = 64, tile_dim2 = 0
  g1[5] = K_;                          // tensor_dim0_stride = 832 elements
  g1[6] = 0;
  g1[7] = 0;
  i32x4 z4 = {0, 0, 0, 0};
#if TDM_ARITY == 6
  i32x8 z8 = {0, 0, 0, 0, 0, 0, 0, 0};
  __builtin_amdgcn_tensor_load_to_lds(g0, g1, z4, z4, z8, 0);
#else
  __builtin_amdgcn_tensor_load_to_lds(g0, g1, z4, z4, 0);
#endif
}
#endif

// Fallback per-lane staging: BT*KC/8 = 512 b128 transfers -> 2 per thread.
__device__ __forceinline__ void stage_chunk(const _Float16* __restrict__ gsrc,
                                            _Float16* lds_dst, int tid) {
#pragma unroll
  for (int it = 0; it < 2; ++it) {
    int c   = tid + it * 256;
    int row = c >> 3;            // 0..63
    int k8  = (c & 7) * 8;       // 0..56
#if HAVE_ASYNC_LDS
    __builtin_amdgcn_global_load_async_to_lds_b128(
        (AS1 v4i*)(const void*)(gsrc + (size_t)row * K_ + k8),
        (AS3 v4i*)(void*)(lds_dst + row * SA + k8),
        0, 0);
#else
    *(v8h*)(lds_dst + row * SA + k8) =
        *(const v8h*)(gsrc + (size_t)row * K_ + k8);
#endif
  }
}

__device__ __forceinline__ void stage_join(int wid) {
#if HAVE_TDM
  if (wid == 0) __builtin_amdgcn_s_wait_tensorcnt(0);
#elif HAVE_ASYNC_LDS
  __builtin_amdgcn_s_wait_asynccnt(0);
#endif
  (void)wid;
  __syncthreads();
}

__global__ __launch_bounds__(256) void lif_hidden(
    const _Float16* __restrict__ xT,    // (250, 256, 832)
    const _Float16* __restrict__ W1T,   // (1024, 832)
    _Float16* __restrict__ sH)          // (250, 256, 1024)
{
  __shared__ _Float16 Ws[HT * SB];      // W tile, resident for all 250 steps
  __shared__ _Float16 As[2][BT * SA];   // x_t K-chunk, double buffered

  const int tid   = threadIdx.x;
  const int lane  = tid & 31;
  const int wid   = tid >> 5;
  const int waveM = wid >> 1;           // 0..3 -> 16-row group
  const int waveN = wid & 1;            // 0..1 -> 32-col group
  const int b0    = blockIdx.x * BT;    // 4 batch tiles
  const int h0    = blockIdx.y * HT;    // 16 hidden tiles

  // stage the loop-invariant W1T tile into LDS once (async-to-LDS path)
  for (int c = tid; c < HT * (K_ / 8); c += 256) {
    int col = c / (K_ / 8);
    int k8  = (c % (K_ / 8)) * 8;
#if HAVE_ASYNC_LDS
    __builtin_amdgcn_global_load_async_to_lds_b128(
        (AS1 v4i*)(const void*)&W1T[(size_t)(h0 + col) * K_ + k8],
        (AS3 v4i*)(void*)&Ws[col * SB + k8], 0, 0);
#else
    *(v8h*)&Ws[col * SB + k8] = *(const v8h*)&W1T[(size_t)(h0 + col) * K_ + k8];
#endif
  }
#if HAVE_ASYNC_LDS
  __builtin_amdgcn_s_wait_asynccnt(0);
#endif
  __syncthreads();

#if HAVE_TDM
  const unsigned asAddr0 = lds_addr_of(&As[0][0]);
  const unsigned asAddr1 = lds_addr_of(&As[1][0]);
#endif

  // membrane potential carried in accumulator-layout registers
  v8f vh[2];
  vh[0] = {};
  vh[1] = {};

  const int aRow = waveM * 16 + (lane & 15);
  const int aKlo = (lane < 16) ? 0 : 16;     // A frag: 16 contiguous K per lane
  const int bKlo = (lane < 16) ? 0 : 16;     // B frag: 16 contiguous K per lane
  const int mAdd = (lane >> 4) << 3;         // C/D: lanes 16..31 hold M = r+8

  for (int t = 0; t < T_; ++t) {
    const _Float16* xrow = xT + (size_t)t * B_ * K_ + (size_t)b0 * K_;

    // stage K-chunk 0; drop an L2 prefetch hint for next timestep's tile
#if HAVE_TDM
    if (wid == 0) tdm_stage_chunk(xrow, asAddr0);
#else
    stage_chunk(xrow, &As[0][0], tid);
#endif
    if (t + 1 < T_) {
      int row = tid >> 3;
      int k8  = (tid & 7) * 8;
      __builtin_prefetch((const void*)(xrow + (size_t)B_ * K_ +
                                       (size_t)row * K_ + k8), 0, 3);
    }
    stage_join(wid);

    v8f acc[2];
    acc[0] = {};
    acc[1] = {};

    int buf = 0;
#pragma unroll 1
    for (int kb = 0; kb < NKB; ++kb) {
      if (kb + 1 < NKB) {               // prefetch next K-chunk (other buffer)
#if HAVE_TDM
        if (wid == 0)
          tdm_stage_chunk(xrow + (kb + 1) * KC, buf ? asAddr0 : asAddr1);
#else
        stage_chunk(xrow + (kb + 1) * KC, &As[buf ^ 1][0], tid);
#endif
      }

      Frag a[2], bf[2][2];
#pragma unroll
      for (int kk = 0; kk < 2; ++kk) {  // two 32-wide K slices in this chunk
        int kb32 = kk * 32;
        a[kk].h[0] = *(const v8h*)&As[buf][aRow * SA + kb32 + aKlo];
        a[kk].h[1] = *(const v8h*)&As[buf][aRow * SA + kb32 + aKlo + 8];
        int col = waveN * 32 + (lane & 15);
        int kw  = kb * KC + kb32 + bKlo;
        bf[0][kk].h[0] = *(const v8h*)&Ws[col * SB + kw];
        bf[0][kk].h[1] = *(const v8h*)&Ws[col * SB + kw + 8];
        bf[1][kk].h[0] = *(const v8h*)&Ws[(col + 16) * SB + kw];
        bf[1][kk].h[1] = *(const v8h*)&Ws[(col + 16) * SB + kw + 8];
      }
#pragma unroll
      for (int ni = 0; ni < 2; ++ni) {
        acc[ni] = __builtin_amdgcn_wmma_f32_16x16x32_f16(
            false, a[0].v, false, bf[ni][0].v, (short)0, acc[ni], false, false);
        acc[ni] = __builtin_amdgcn_wmma_f32_16x16x32_f16(
            false, a[1].v, false, bf[ni][1].v, (short)0, acc[ni], false, false);
      }
      stage_join(wid);
      buf ^= 1;
    }

    // LIF update + spike emit (f16, in kernel B's A-fragment-friendly layout)
    _Float16* srow = sH + (size_t)t * B_ * H_;
#pragma unroll
    for (int ni = 0; ni < 2; ++ni) {
      int colG = h0 + waveN * 32 + ni * 16 + (lane & 15);
#pragma unroll
      for (int r = 0; r < 8; ++r) {
        float v = vh[ni][r] + 0.05f * (0.0f - vh[ni][r]) + acc[ni][r];
        float s = (v >= 1.0f) ? 1.0f : 0.0f;
        vh[ni][r] = (s > 0.0f) ? 0.0f : v;
        int rowG = b0 + waveM * 16 + r + mAdd;
        srow[(size_t)rowG * H_ + colG] = (_Float16)s;
      }
    }
  }
}

// ------------------------- output layer ------------------------------------

__global__ __launch_bounds__(128) void lif_output(
    const _Float16* __restrict__ sH,    // (250, 256, 1024)
    const _Float16* __restrict__ W2T,   // (16, 1024)
    float* __restrict__ out)            // spikes (256,10,250) ++ rates (256,10)
{
  const int tid   = threadIdx.x;
  const int lane  = tid & 31;
  const int w     = blockIdx.x * 4 + (tid >> 5);   // 0..15 -> 16-batch tile
  const int bBase = w * 16;
  const int o     = lane & 15;
  const int mAdd  = (lane >> 4) << 3;
  const int aKlo  = (lane < 16) ? 0 : 16;
  const int bKlo  = (lane < 16) ? 0 : 16;

  v8f vo = {};
  v8f fr = {};

  // t = 0 uses s_h_prev == 0 -> no output spikes
  if (o < NOUT) {
#pragma unroll
    for (int r = 0; r < 8; ++r)
      out[(size_t)(bBase + r + mAdd) * (NOUT * T_) + o * T_ + 0] = 0.0f;
  }

#pragma unroll 1
  for (int t = 1; t < T_; ++t) {
    const _Float16* sprev = sH + (size_t)(t - 1) * B_ * H_;
    v8f acc = {};
#pragma unroll 2
    for (int kb = 0; kb < H_ / 32; ++kb) {
      Frag a, bf;
      int row = bBase + (lane & 15);
      int ka  = kb * 32 + aKlo;
      a.h[0] = *(const v8h*)&sprev[(size_t)row * H_ + ka];
      a.h[1] = *(const v8h*)&sprev[(size_t)row * H_ + ka + 8];
      int kbK = kb * 32 + bKlo;
      bf.h[0] = *(const v8h*)&W2T[(size_t)o * H_ + kbK];
      bf.h[1] = *(const v8h*)&W2T[(size_t)o * H_ + kbK + 8];
      acc = __builtin_amdgcn_wmma_f32_16x16x32_f16(
          false, a.v, false, bf.v, (short)0, acc, false, false);
    }
#pragma unroll
    for (int r = 0; r < 8; ++r) {
      float v = vo[r] + 0.05f * (0.0f - vo[r]) + acc[r];
      float s = (v >= 1.0f) ? 1.0f : 0.0f;
      vo[r] = (s > 0.0f) ? 0.0f : v;
      fr[r] += s;
      if (o < NOUT)
        out[(size_t)(bBase + r + mAdd) * (NOUT * T_) + o * T_ + t] = s;
    }
  }

  if (o < NOUT) {
#pragma unroll
    for (int r = 0; r < 8; ++r)
      out[(size_t)B_ * NOUT * T_ + (bBase + r + mAdd) * NOUT + o] =
          fr[r] * (1.0f / T_);
  }
}

// ------------------------- launcher ----------------------------------------

extern "C" void kernel_launch(void* const* d_in, const int* in_sizes, int n_in,
                              void* d_out, int out_size, void* d_ws, size_t ws_size,
                              hipStream_t stream) {
  (void)in_sizes; (void)n_in; (void)out_size; (void)ws_size;

  const float* x  = (const float*)d_in[0];   // (256, 784, 250)
  const float* w1 = (const float*)d_in[1];   // (784, 1024)
  const float* w2 = (const float*)d_in[2];   // (1024, 10)
  float* out = (float*)d_out;

  char*  ws  = (char*)d_ws;
  size_t off = 0;
  auto wsAlloc = [&](size_t bytes) -> void* {
    void* p = (void*)(ws + off);
    off += (bytes + 255) & ~(size_t)255;
    return p;
  };
  _Float16* xT  = (_Float16*)wsAlloc((size_t)T_ * B_ * K_ * sizeof(_Float16));
  _Float16* W1T = (_Float16*)wsAlloc((size_t)H_ * K_ * sizeof(_Float16));
  _Float16* W2T = (_Float16*)wsAlloc((size_t)16 * H_ * sizeof(_Float16));
  _Float16* sHs = (_Float16*)wsAlloc((size_t)T_ * B_ * H_ * sizeof(_Float16));

  transpose_input<<<dim3(B_, K_ / 32, 8), 256, 0, stream>>>(x, xT);
  transpose_w1<<<dim3(K_ / 32, H_ / 32), 256, 0, stream>>>(w1, W1T);
  prep_w2<<<dim3((16 * H_ + 255) / 256), 256, 0, stream>>>(w2, W2T);
  lif_hidden<<<dim3(B_ / BT, H_ / HT), 256, 0, stream>>>(xT, W1T, sHs);
  lif_output<<<dim3(4), 128, 0, stream>>>(sHs, W2T, out);
}
